// UpsampleConv_71554155151951
// MI455X (gfx1250) — matrix operand order repaired and checked by reading, not compiled, not run on hardware
//
#include <hip/hip_runtime.h>

// Problem constants (from reference)
#define BB   8
#define CIN  128
#define HH   64
#define WW   64
#define SS   8
#define COUT 128

typedef __attribute__((ext_vector_type(16))) __bf16 v16bf;
typedef __attribute__((ext_vector_type(8)))  float  v8f;
typedef int int4v __attribute__((ext_vector_type(4)));

struct alignas(16) Q16 { unsigned int a, b, c, d; }; // 16-byte chunk
union Frag16 { v16bf v; Q16 q[2]; };

// gfx1250 async global->LDS staging if the toolchain exposes it
#if defined(__has_builtin)
#  if __has_builtin(__builtin_amdgcn_global_load_async_to_lds_b128)
#    define USE_ASYNC_LDS 1
#  endif
#endif
#ifndef USE_ASYNC_LDS
#  define USE_ASYNC_LDS 0
#endif

#if defined(__has_builtin)
#  if __has_builtin(__builtin_amdgcn_s_wait_asynccnt)
#    define WAIT_ASYNC() __builtin_amdgcn_s_wait_asynccnt(0)
#  endif
#endif
#ifndef WAIT_ASYNC
#  define WAIT_ASYNC() asm volatile("s_wait_asynccnt 0x0" ::: "memory")
#endif

// Typed, address-space-qualified pointers matching the builtin signature:
//   param0: int4 in AS1 (global / __device__), param1: int4 in AS3 (__shared__)
typedef __attribute__((address_space(1))) int4v* gptr_b128;
typedef __attribute__((address_space(3))) int4v* lptr_b128;
#define AS1P(p) ((gptr_b128)(p))
#define AS3P(p) ((lptr_b128)(p))

// Workspace layout:
//   xT   : [B][66][66][CIN]        bf16  (zero-padded halo, channels-last)
//   Weff : [4][S][COUT][4][CIN]    bf16  (per-parity folded 2x2 kernels)
//   sel  : [B][S][64*64]           f32   (softmax coupler, source grid)
#define XT_ELEMS   (BB * 66 * 66 * CIN)
#define WEFF_ELEMS (4 * SS * COUT * 4 * CIN)
#define SEL_ELEMS  (BB * SS * HH * WW)

// ---------------------------------------------------------------------------
// Prep 1: fp32 NCHW -> bf16 padded channels-last
// ---------------------------------------------------------------------------
__global__ void UpsampleConv_prep_x(const float* __restrict__ x,
                                    __bf16* __restrict__ xT) {
    int idx = blockIdx.x * blockDim.x + threadIdx.x;
    if (idx >= XT_ELEMS) return;
    int c  = idx & (CIN - 1);
    int t  = idx >> 7;
    int jj = t % 66; t /= 66;
    int ii = t % 66;
    int b  = t / 66;
    float v = 0.f;
    if (ii >= 1 && ii <= HH && jj >= 1 && jj <= WW)
        v = x[((b * CIN + c) * HH + (ii - 1)) * WW + (jj - 1)];
    xT[idx] = (__bf16)v;
}

// ---------------------------------------------------------------------------
// Prep 2: fold 3x3 kernels into per-parity effective 2x2 kernels (bf16)
//   Row tap sets: py=0: ty0={kh0}, ty1={kh1,kh2}; py=1: ty0={kh0,kh1}, ty1={kh2}
// ---------------------------------------------------------------------------
__global__ void UpsampleConv_prep_w(const float* __restrict__ W,
                                    __bf16* __restrict__ Weff) {
    int idx = blockIdx.x * blockDim.x + threadIdx.x;
    if (idx >= WEFF_ELEMS) return;
    int c   = idx & (CIN - 1);
    int t   = idx >> 7;
    int tap = t & 3;  t >>= 2;
    int o   = t & (COUT - 1); t >>= 7;
    int s   = t & (SS - 1);
    int p   = t >> 3;
    int py = p >> 1, px = p & 1;
    int ty = tap >> 1, tx = tap & 1;
    int lo_r = ty * (1 + py);
    int hi_r = ty ? 2 : py;
    int lo_c = tx * (1 + px);
    int hi_c = tx ? 2 : px;
    float sum = 0.f;
    for (int kh = lo_r; kh <= hi_r; ++kh)
        for (int kw = lo_c; kw <= hi_c; ++kw)
            sum += W[(((s * COUT + o) * CIN + c) * 3 + kh) * 3 + kw];
    Weff[idx] = (__bf16)sum;
}

// ---------------------------------------------------------------------------
// Prep 3: coupler softmax on the SOURCE grid (identical for all 4 sub-pixels)
// ---------------------------------------------------------------------------
__global__ void UpsampleConv_prep_sel(const float* __restrict__ x,
                                      const float* __restrict__ Wc,
                                      const float* __restrict__ bc,
                                      float* __restrict__ sel) {
    int idx = blockIdx.x * blockDim.x + threadIdx.x;
    if (idx >= BB * HH * WW) return;
    const int j = idx & 63;
    const int i = (idx >> 6) & 63;
    const int b = idx >> 12;
    float logit[SS];
#pragma unroll
    for (int s = 0; s < SS; ++s) logit[s] = bc[s];
    for (int c = 0; c < CIN; ++c) {
        const float xv = x[((b * CIN + c) * HH + i) * WW + j];
#pragma unroll
        for (int s = 0; s < SS; ++s) logit[s] += xv * Wc[s * CIN + c];
    }
    float mx = logit[0];
#pragma unroll
    for (int s = 1; s < SS; ++s) mx = fmaxf(mx, logit[s]);
    float sum = 0.f;
#pragma unroll
    for (int s = 0; s < SS; ++s) { logit[s] = __expf(logit[s] - mx); sum += logit[s]; }
    const float inv = 1.f / sum;
#pragma unroll
    for (int s = 0; s < SS; ++s)
        sel[((b * SS + s) * HH + i) * WW + j] = logit[s] * inv;
}

// ---------------------------------------------------------------------------
// Main: implicit GEMM over folded 2x2 taps, double-buffered LDS B staging.
//   block = 256 threads = 8 waves; block covers 8 pixel tiles (2 source rows
//   x 4 col-tiles of 16) for ONE 16-wide out-channel tile of one parity.
//   Wave: 16 pixels x 16 oc, K = 4 taps * 128 cin = 512, WMMA bf16 16x16x32.
//   B slice (8 s * 16 oc * 32 cin bf16 = 8KB) is async-DMA'd into LDS for
//   step k+1 while step k computes; all 8 B fragments are preloaded into
//   registers before the 8-WMMA chain.
// ---------------------------------------------------------------------------
__global__ __launch_bounds__(256) void UpsampleConv_main(
    const __bf16* __restrict__ xT, const __bf16* __restrict__ Weff,
    const float* __restrict__ sel, const float* __restrict__ bias,
    float* __restrict__ out) {
    __shared__ __bf16 ldsB[2][SS * 16 * 32];   // [buf][s][oc16][cin32] : 2 x 8 KB

    const int ipair = blockIdx.x & 31;      // source-row pair
    const int ocb   = blockIdx.x >> 5;      // out-channel tile 0..7
    const int b     = blockIdx.y;
    const int p     = blockIdx.z;           // parity
    const int py = p >> 1, px = p & 1;

    const int tid   = threadIdx.x;
    const int wave  = tid >> 5;
    const int lane  = tid & 31;
    const int m     = lane & 15;            // A row / D column index
    const int khalf = lane >> 4;            // K-half select

    const int i      = ipair * 2 + (wave >> 2);  // source row
    const int j_base = (wave & 3) * 16;          // source col tile base

    // LDS-fill mapping: 256 threads x 32B == 8s * 16oc * 64B rows (8 KB)
    const int fs    = tid >> 5;
    const int fr    = tid & 31;
    const int foc   = fr >> 1;
    const int fhalf = fr & 1;               // which 32B half of the 64B row
    const int fdst  = (fs * 16 + foc) * 32 + fhalf * 16;  // elements
    const __bf16* fsrc_base =
        Weff + ((((p * SS + fs) * COUT + (ocb * 16 + foc)) * 4) * CIN + fhalf * 16);

    // issue the B-slice fill for K-step kk into buffer `buf`
    auto issue_fill = [&](int kk, int buf) {
        const int tap = kk >> 2;
        const int c0  = (kk & 3) * 32;
        const __bf16* src = fsrc_base + tap * CIN + c0;   // 32B contiguous
        __bf16* dst = &ldsB[buf][fdst];
#if USE_ASYNC_LDS
        __builtin_amdgcn_global_load_async_to_lds_b128(AS1P(src),     AS3P(dst),     0, 0);
        __builtin_amdgcn_global_load_async_to_lds_b128(AS1P(src + 8), AS3P(dst + 8), 0, 0);
#else
        ((Q16*)dst)[0] = ((const Q16*)src)[0];
        ((Q16*)dst)[1] = ((const Q16*)src)[1];
#endif
    };

    const v8f vzero = {0.f, 0.f, 0.f, 0.f, 0.f, 0.f, 0.f, 0.f};
    v8f acc[SS];
#pragma unroll
    for (int s = 0; s < SS; ++s) acc[s] = vzero;

    // prologue: stage slice 0
    issue_fill(0, 0);
#if USE_ASYNC_LDS
    WAIT_ASYNC();
#endif
    __syncthreads();

    for (int kk = 0; kk < 16; ++kk) {       // 4 taps * (128/32) cin chunks
        const int cur = kk & 1;
        if (kk + 1 < 16) issue_fill(kk + 1, cur ^ 1);

        const int tap = kk >> 2;
        const int c0  = (kk & 3) * 32;
        const int ty = tap >> 1, tx = tap & 1;

        // A fragment: 16-bit A 16x32 layout -> two 16B contiguous loads
        Frag16 fa;
        {
            const __bf16* pa = xT +
                (((b * 66) + (i + ty + py)) * 66 + (j_base + m + tx + px)) * CIN +
                c0 + khalf * 8;
            fa.q[0] = *(const Q16*)pa;
            fa.q[1] = *(const Q16*)(pa + 16);
        }

        // preload all 8 B fragments, then run the WMMA chain back-to-back
        Frag16 fb[SS];
        const __bf16* bufc = &ldsB[cur][0];
#pragma unroll
        for (int s = 0; s < SS; ++s) {
            const __bf16* pb = bufc + (s * 16 + m) * 32 + khalf * 8;
            fb[s].q[0] = *(const Q16*)pb;
            fb[s].q[1] = *(const Q16*)(pb + 16);
        }
#pragma unroll
        for (int s = 0; s < SS; ++s) {
            acc[s] = __builtin_amdgcn_wmma_f32_16x16x32_bf16(
                false, fa.v, false, fb[s].v, (short)0, acc[s], false, false);
        }

        if (kk + 1 < 16) {
#if USE_ASYNC_LDS
            WAIT_ASYNC();                   // next slice landed in LDS
#endif
            __syncthreads();                // all waves done reading `cur`
        }
    }

    // Epilogue: out[m][n] = sum_s sel_s[m] * (acc_s[m][n] + bias[s][n])
    const int o = ocb * 16 + m;             // D column -> out channel
    float fin[8] = {0, 0, 0, 0, 0, 0, 0, 0};
#pragma unroll
    for (int s = 0; s < SS; ++s) {
        const float bs = bias[s * COUT + o];
        const float* sp = sel + (((b * SS + s) * HH + i) * WW + j_base + khalf * 8);
        const float4 s0 = *(const float4*)sp;
        const float4 s1 = *(const float4*)(sp + 4);
        const float sv[8] = {s0.x, s0.y, s0.z, s0.w, s1.x, s1.y, s1.z, s1.w};
#pragma unroll
        for (int r = 0; r < 8; ++r)
            fin[r] += sv[r] * (acc[s][r] + bs);
    }
#pragma unroll
    for (int r = 0; r < 8; ++r) {
        const int j = j_base + khalf * 8 + r;       // D row -> source col
        out[((b * COUT + o) * (2 * HH) + (2 * i + py)) * (2 * WW) + (2 * j + px)] =
            fin[r];
    }
}

// ---------------------------------------------------------------------------
extern "C" void kernel_launch(void* const* d_in, const int* in_sizes, int n_in,
                              void* d_out, int out_size, void* d_ws, size_t ws_size,
                              hipStream_t stream) {
    const float* x  = (const float*)d_in[0];  // [8,128,64,64]
    const float* W  = (const float*)d_in[1];  // [8,128,128,3,3]
    const float* bb = (const float*)d_in[2];  // [8,128]
    const float* Wc = (const float*)d_in[3];  // [8,128]
    const float* bc = (const float*)d_in[4];  // [8]
    float* out = (float*)d_out;               // [8,128,128,128]

    char* ws = (char*)d_ws;
    __bf16* xT   = (__bf16*)ws;
    __bf16* Weff = (__bf16*)(ws + (size_t)XT_ELEMS * 2);
    float*  sel  = (float*)(ws + (size_t)XT_ELEMS * 2 + (size_t)WEFF_ELEMS * 2);

    UpsampleConv_prep_x<<<(XT_ELEMS + 255) / 256, 256, 0, stream>>>(x, xT);
    UpsampleConv_prep_w<<<(WEFF_ELEMS + 255) / 256, 256, 0, stream>>>(W, Weff);
    UpsampleConv_prep_sel<<<(BB * HH * WW + 255) / 256, 256, 0, stream>>>(x, Wc, bc, sel);

    dim3 grid(256 /* 32 row-pairs * 8 oc tiles */, BB, 4 /* parity */);
    UpsampleConv_main<<<grid, 256, 0, stream>>>(xT, Weff, sel, bb, out);
}
